// GatedGCNNet_21079699489187
// MI455X (gfx1250) — compile-verified, matrix operand order checked by reading, not run on hardware
//
#include <hip/hip_runtime.h>
#include <hip/hip_bf16.h>

typedef __attribute__((ext_vector_type(16))) _Float16 v16h;
typedef __attribute__((ext_vector_type(8)))  float    v8f;

#define NNODES 50000
#define NEDGES 800000

// ---------------------------------------------------------------------------
// WMMA helper: D = A(16x32 f16) * B(32x16 f16) + C(f32)
// ---------------------------------------------------------------------------
__device__ __forceinline__ v8f wmma_f16(v16h a, v16h b, v8f c) {
  return __builtin_amdgcn_wmma_f32_16x16x32_f16(
      /*neg_a=*/false, a, /*neg_b=*/false, b,
      /*c_mod=*/(short)0, c, /*reuse_a=*/false, /*reuse_b=*/false);
}

// Agent-scope native f32 atomic add (global_atomic_add_f32)
__device__ __forceinline__ void gatomic_add(float* p, float v) {
  __hip_atomic_fetch_add(p, v, __ATOMIC_RELAXED, __HIP_MEMORY_SCOPE_AGENT);
}

// Fast reciprocal: single v_rcp_f32 instead of IEEE div sequence.
__device__ __forceinline__ float fast_rcp(float x) {
  return __builtin_amdgcn_rcpf(x);
}

// ---------------------------------------------------------------------------
// A-fragment loader. ISA 16-bit A 16x32 layout:
//   lane L: m = L&15, g = L>>4
//   halves 0..7  = A[m][kbase + g*8 + 0..7]
//   halves 8..15 = A[m][kbase + 16 + g*8 + 0..7]
// KREAL (8/16/64) zero-pads out-of-range K (for the K<32 embedding GEMMs).
// ---------------------------------------------------------------------------
template<int KREAL>
__device__ __forceinline__ v16h load_afrag(const float* __restrict__ row,
                                           int kbase, int g) {
  v16h a;
#pragma unroll
  for (int i = 0; i < 16; ++i) {
    int k = kbase + ((i & 8) ? (16 + g * 8 + (i & 7)) : (g * 8 + (i & 7)));
    float v = 0.f;
    if (KREAL >= 64) v = row[k];
    else if (k < KREAL) v = row[k];
    a[i] = (_Float16)v;
  }
  return a;
}

// Packed B-fragment: [frag][lane][16 halves], 1KB per fragment, coalesced.
__device__ __forceinline__ v16h load_bfrag(const _Float16* __restrict__ wf,
                                           int frag, int lane) {
  return *(const v16h*)(wf + (size_t)frag * 512 + lane * 16);
}

// ---------------------------------------------------------------------------
// Weight pre-pack: f32 W[K x Nout] (row-major) -> f16 B-fragments.
// frag = s*NT + nt ; within frag: lane L holds column n = nt*16 + (L&15),
// halves map to K exactly mirroring the A-fragment K mapping.
// ---------------------------------------------------------------------------
__global__ void k_pack_w(const float* __restrict__ W, int K, int Nout,
                         _Float16* __restrict__ dst) {
  int S = (K + 31) / 32;
  int NT = Nout / 16;
  int total = S * NT * 512;
  int idx = blockIdx.x * 256 + threadIdx.x;
  if (idx >= total) return;
  int hi   = idx & 15;
  int lane = (idx >> 4) & 31;
  int frag = idx >> 9;
  int nt = frag % NT;
  int s  = frag / NT;
  int g  = lane >> 4;
  int n  = nt * 16 + (lane & 15);
  int k  = s * 32 + ((hi & 8) ? (16 + g * 8 + (hi & 7)) : (g * 8 + (hi & 7)));
  float v = (k < K) ? W[(size_t)k * Nout + n] : 0.f;
  dst[idx] = (_Float16)v;
}

__global__ void k_zero(float* __restrict__ p, size_t count) {
  size_t i = (size_t)blockIdx.x * 256 + threadIdx.x;
  if (i < count) p[i] = 0.f;
}

// ---------------------------------------------------------------------------
// Generic WMMA GEMM: out[m, colOff + 0..63] = X[M x KREAL] @ Wpk + bias
// (+ optional addSrc).  One wave = 16 rows x 64 cols, KREAL via S k-steps.
// ---------------------------------------------------------------------------
template<int KREAL, int S>
__global__ __launch_bounds__(256) void k_gemm_tile(
    const float* __restrict__ X, int ldX, int M,
    const _Float16* __restrict__ wf, const float* __restrict__ bias,
    const float* __restrict__ addSrc,
    float* __restrict__ out, int ldOut, int colOff) {
  const int lane  = threadIdx.x & 31;
  const int wave  = threadIdx.x >> 5;
  const int mtile = blockIdx.x * 8 + wave;
  if (mtile * 16 >= M) return;            // wave-uniform: EXEC stays all-ones
  const int g = lane >> 4;
  const int m = mtile * 16 + (lane & 15);

  v16h a[S];
#pragma unroll
  for (int s = 0; s < S; ++s)
    a[s] = load_afrag<KREAL>(X + (size_t)m * ldX, 32 * s, g);

#pragma unroll
  for (int nt = 0; nt < 4; ++nt) {
    v8f acc = {};
#pragma unroll
    for (int s = 0; s < S; ++s)
      acc = wmma_f16(a[s], load_bfrag(wf, s * 4 + nt, lane), acc);
    const int nL = nt * 16 + (lane & 15);
    const float bv = bias[nL];
#pragma unroll
    for (int r = 0; r < 8; ++r) {
      int mr = mtile * 16 + 8 * g + r;
      float v = acc[r] + bv;
      if (addSrc) v += addSrc[(size_t)mr * ldOut + colOff + nL];
      out[(size_t)mr * ldOut + colOff + nL] = v;
    }
  }
}

// ---------------------------------------------------------------------------
// Fused edge kernel: Ce = e @ Wc + bc (WMMA), e_new = Ce + Dh[src] + Eh[dst],
// sigma = sigmoid(e_new), atomics into num/den (segment sum over dst),
// LDS-reduced BN statistics for e_new.  nodeLin = [N x 256] (A|B|D|E).
// NE/16 tiles divide blockDim*8 exactly -> no tail, safe __syncthreads.
// ---------------------------------------------------------------------------
__global__ __launch_bounds__(256) void k_edge(
    const float* __restrict__ e, const _Float16* __restrict__ wf,
    const float* __restrict__ biasC, const float* __restrict__ nodeLin,
    const int* __restrict__ src, const int* __restrict__ dst,
    float* __restrict__ eNew, float* __restrict__ num, float* __restrict__ den,
    float* __restrict__ eSum, float* __restrict__ eSq) {
  __shared__ float sSum[64];
  __shared__ float sSq[64];
  if (threadIdx.x < 64) { sSum[threadIdx.x] = 0.f; sSq[threadIdx.x] = 0.f; }
  __syncthreads();

  const int lane  = threadIdx.x & 31;
  const int wave  = threadIdx.x >> 5;
  const int mtile = blockIdx.x * 8 + wave;
  const int g = lane >> 4;
  const int m = mtile * 16 + (lane & 15);

  int si[8], di[8];
#pragma unroll
  for (int r = 0; r < 8; ++r) {
    int mE = mtile * 16 + 8 * g + r;
    si[r] = src[mE];
    di[r] = dst[mE];
  }
  // Overlap the random node-row gathers with the WMMA work.
#pragma unroll
  for (int r = 0; r < 8; ++r) {
    __builtin_prefetch(nodeLin + (size_t)si[r] * 256 + 64, 0, 1);   // Bh,Dh
    __builtin_prefetch(nodeLin + (size_t)di[r] * 256 + 192, 0, 1);  // Eh
  }

  v16h a0 = load_afrag<64>(e + (size_t)m * 64, 0, g);
  v16h a1 = load_afrag<64>(e + (size_t)m * 64, 32, g);

#pragma unroll
  for (int nt = 0; nt < 4; ++nt) {
    v8f acc = {};
    acc = wmma_f16(a0, load_bfrag(wf, nt, lane), acc);
    acc = wmma_f16(a1, load_bfrag(wf, 4 + nt, lane), acc);
    const int nL = nt * 16 + (lane & 15);
    const float bv = biasC[nL];
#pragma unroll
    for (int r = 0; r < 8; ++r) {
      int mE = mtile * 16 + 8 * g + r;
      float Dv = nodeLin[(size_t)si[r] * 256 + 128 + nL];
      float Ev = nodeLin[(size_t)di[r] * 256 + 192 + nL];
      float en = acc[r] + bv + Dv + Ev;
      eNew[(size_t)mE * 64 + nL] = en;
      // fast sigmoid: v_exp_f32 + v_rcp_f32 (avoids IEEE div sequence)
      float sg = fast_rcp(1.f + __expf(-en));
      float Bv = nodeLin[(size_t)si[r] * 256 + 64 + nL];
      gatomic_add(&num[(size_t)di[r] * 64 + nL], sg * Bv);
      gatomic_add(&den[(size_t)di[r] * 64 + nL], sg);
      atomicAdd(&sSum[nL], en);        // ds_add_f32
      atomicAdd(&sSq[nL], en * en);
    }
  }
  __syncthreads();
  if (threadIdx.x < 64) {
    gatomic_add(&eSum[threadIdx.x], sSum[threadIdx.x]);
    gatomic_add(&eSq[threadIdx.x],  sSq[threadIdx.x]);
  }
}

// ---------------------------------------------------------------------------
// h_new = Ah + num / (den + 1e-6); accumulate BN stats (LDS reduce -> global)
// total = N*64 (exact multiple of 256)
// ---------------------------------------------------------------------------
__global__ __launch_bounds__(256) void k_hcombine(
    const float* __restrict__ nodeLin, const float* __restrict__ num,
    const float* __restrict__ den, float* __restrict__ hNew,
    float* __restrict__ hSum, float* __restrict__ hSq, int total) {
  __shared__ float sSum[64];
  __shared__ float sSq[64];
  if (threadIdx.x < 64) { sSum[threadIdx.x] = 0.f; sSq[threadIdx.x] = 0.f; }
  __syncthreads();
  int idx = blockIdx.x * 256 + threadIdx.x;
  if (idx < total) {
    int mN = idx >> 6, n = idx & 63;
    float v = nodeLin[(size_t)mN * 256 + n] +
              num[idx] * fast_rcp(den[idx] + 1e-6f);
    hNew[idx] = v;
    atomicAdd(&sSum[n], v);
    atomicAdd(&sSq[n], v * v);
  }
  __syncthreads();
  if (threadIdx.x < 64) {
    gatomic_add(&hSum[threadIdx.x], sSum[threadIdx.x]);
    gatomic_add(&hSq[threadIdx.x],  sSq[threadIdx.x]);
  }
}

// out = base + relu(gamma * (x - mu) * rsqrt(var + eps) + beta), in place
__global__ __launch_bounds__(256) void k_bn_apply(
    const float* __restrict__ xNew, const float* __restrict__ sum,
    const float* __restrict__ sq, const float* __restrict__ gamma,
    const float* __restrict__ beta, float* __restrict__ inout,
    float invM, size_t total) {
  size_t idx = (size_t)blockIdx.x * 256 + threadIdx.x;
  if (idx >= total) return;
  int n = idx & 63;
  float mu  = sum[n] * invM;
  float var = sq[n] * invM - mu * mu;
  float v = gamma[n] * (xNew[idx] - mu) * rsqrtf(var + 1e-5f) + beta[n];
  inout[idx] += fmaxf(v, 0.f);
}

// ---------------------------------------------------------------------------
// MLP readout 64->32->16->10, weights cached in LDS, one thread per node.
// ---------------------------------------------------------------------------
__global__ __launch_bounds__(256) void k_readout(
    const float* __restrict__ h,
    const float* __restrict__ W0, const float* __restrict__ b0,
    const float* __restrict__ W1, const float* __restrict__ b1,
    const float* __restrict__ W2, const float* __restrict__ b2,
    float* __restrict__ out, int Nn) {
  __shared__ float sW0[64 * 32];
  __shared__ float sW1[32 * 16];
  __shared__ float sW2[16 * 10];
  __shared__ float sb0[32], sb1[16], sb2[10];
  for (int i = threadIdx.x; i < 2048; i += 256) sW0[i] = W0[i];
  for (int i = threadIdx.x; i < 512;  i += 256) sW1[i] = W1[i];
  for (int i = threadIdx.x; i < 160;  i += 256) sW2[i] = W2[i];
  if (threadIdx.x < 32) sb0[threadIdx.x] = b0[threadIdx.x];
  if (threadIdx.x < 16) sb1[threadIdx.x] = b1[threadIdx.x];
  if (threadIdx.x < 10) sb2[threadIdx.x] = b2[threadIdx.x];
  __syncthreads();
  int node = blockIdx.x * 256 + threadIdx.x;
  if (node >= Nn) return;
  float hv[64];
  const float4* hp = (const float4*)(h + (size_t)node * 64);
#pragma unroll
  for (int i = 0; i < 16; ++i) {
    float4 v = hp[i];
    hv[4 * i] = v.x; hv[4 * i + 1] = v.y; hv[4 * i + 2] = v.z; hv[4 * i + 3] = v.w;
  }
  float y0[32];
#pragma unroll 4
  for (int j = 0; j < 32; ++j) {
    float a = sb0[j];
    for (int k = 0; k < 64; ++k) a += hv[k] * sW0[k * 32 + j];
    y0[j] = fmaxf(a, 0.f);
  }
  float y1[16];
#pragma unroll 4
  for (int j = 0; j < 16; ++j) {
    float a = sb1[j];
    for (int k = 0; k < 32; ++k) a += y0[k] * sW1[k * 16 + j];
    y1[j] = fmaxf(a, 0.f);
  }
#pragma unroll
  for (int j = 0; j < 10; ++j) {
    float a = sb2[j];
    for (int k = 0; k < 16; ++k) a += y1[k] * sW2[k * 10 + j];
    out[(size_t)node * 10 + j] = a;
  }
}

// ---------------------------------------------------------------------------
// Host-side orchestration
// ---------------------------------------------------------------------------
extern "C" void kernel_launch(void* const* d_in, const int* in_sizes, int n_in,
                              void* d_out, int out_size, void* d_ws, size_t ws_size,
                              hipStream_t stream) {
  (void)in_sizes; (void)n_in; (void)out_size; (void)ws_size;
  const float* h_in   = (const float*)d_in[0];
  const float* e_feat = (const float*)d_in[1];
  const float* pos    = (const float*)d_in[2];
  const int*   src    = (const int*)d_in[3];
  const int*   dst    = (const int*)d_in[4];
  const float* We     = (const float*)d_in[5];
  const float* be     = (const float*)d_in[6];
  const float* Wp     = (const float*)d_in[7];
  const float* bp     = (const float*)d_in[8];
  const float* linW   = (const float*)d_in[9];
  const float* linb   = (const float*)d_in[10];
  const float* gamma  = (const float*)d_in[11];
  const float* beta   = (const float*)d_in[12];
  const float* W0 = (const float*)d_in[13]; const float* b0 = (const float*)d_in[14];
  const float* W1 = (const float*)d_in[15]; const float* b1 = (const float*)d_in[16];
  const float* W2 = (const float*)d_in[17]; const float* b2 = (const float*)d_in[18];

  // Workspace layout (bytes)
  char* ws = (char*)d_ws;
  _Float16* PW  = (_Float16*)(ws);                        // 168 KB packed f16 weights
  float* H      = (float*)(ws + 172032);                  // N x 64
  float* E      = (float*)(ws + 12972032);                // NE x 64
  float* NL     = (float*)(ws + 217772032);               // N x 256 (A|B|D|E)
  float* ENEW   = (float*)(ws + 268972032);               // NE x 64
  float* HNEW   = (float*)(ws + 473772032);               // N x 64
  float* NUM    = (float*)(ws + 486572032);               // N x 64
  float* DEN    = (float*)(ws + 499372032);               // N x 64
  float* STATS  = (float*)(ws + 512172032);               // hSum|hSq|eSum|eSq (4x64)
  float* HSUM = STATS;       float* HSQ = STATS + 64;
  float* ESUM = STATS + 128; float* ESQ = STATS + 192;

  // ---- 1. pack weights into WMMA B-fragment layout (f16) ----
  k_pack_w<<<8, 256, 0, stream>>>(We, 16, 64, PW);                 // frags 0..3
  k_pack_w<<<8, 256, 0, stream>>>(Wp,  8, 64, PW + 4 * 512);       // frags 4..7
  for (int l = 0; l < 4; ++l)
    for (int j = 0; j < 5; ++j)
      k_pack_w<<<16, 256, 0, stream>>>(linW + (size_t)(l * 5 + j) * 4096, 64, 64,
                                       PW + (size_t)(8 + (l * 5 + j) * 8) * 512);

  // ---- 2. embeddings ----
  // E = e_feat @ We + be        (K=16, zero-padded)
  k_gemm_tile<16, 1><<<6250, 256, 0, stream>>>(e_feat, 16, NEDGES, PW, be,
                                               nullptr, E, 64, 0);
  // H = h_in + pos @ Wp + bp    (K=8, zero-padded)
  k_gemm_tile<8, 1><<<391, 256, 0, stream>>>(pos, 8, NNODES, PW + 4 * 512, bp,
                                             h_in, H, 64, 0);

  // ---- 3. GatedGCN layers ----
  const int jmap[4] = {0, 1, 3, 4};  // A,B,D,E -> nodeLin cols 0,64,128,192
  for (int l = 0; l < 4; ++l) {
    // zero NUM | DEN | STATS (contiguous): 2*N*64 + 256 floats
    k_zero<<<25001, 256, 0, stream>>>(NUM, (size_t)6400256);
    for (int jj = 0; jj < 4; ++jj) {
      int j = jmap[jj];
      k_gemm_tile<64, 2><<<391, 256, 0, stream>>>(
          H, 64, NNODES, PW + (size_t)(8 + (l * 5 + j) * 8) * 512,
          linb + (size_t)(l * 5 + j) * 64, nullptr, NL, 256, jj * 64);
    }
    // fused edge pass: Ce WMMA + gathers + sigmoid + segment-sum + BN stats
    k_edge<<<6250, 256, 0, stream>>>(
        E, PW + (size_t)(8 + (l * 5 + 2) * 8) * 512,
        linb + (size_t)(l * 5 + 2) * 64, NL, src, dst,
        ENEW, NUM, DEN, ESUM, ESQ);
    k_hcombine<<<12500, 256, 0, stream>>>(NL, NUM, DEN, HNEW, HSUM, HSQ,
                                          NNODES * 64);
    k_bn_apply<<<12500, 256, 0, stream>>>(HNEW, HSUM, HSQ,
                                          gamma + l * 128, beta + l * 128,
                                          H, 1.f / NNODES, (size_t)NNODES * 64);
    k_bn_apply<<<200000, 256, 0, stream>>>(ENEW, ESUM, ESQ,
                                           gamma + l * 128 + 64, beta + l * 128 + 64,
                                           E, 1.f / NEDGES, (size_t)NEDGES * 64);
  }

  // ---- 4. readout ----
  k_readout<<<196, 256, 0, stream>>>(H, W0, b0, W1, b1, W2, b2,
                                     (float*)d_out, NNODES);
}
